// BrainAttention_69707319214147
// MI455X (gfx1250) — matrix operand order repaired and checked by reading, not compiled, not run on hardware
//
#include <hip/hip_runtime.h>

typedef __attribute__((ext_vector_type(16))) _Float16 v16h;
typedef __attribute__((ext_vector_type(8)))  _Float16 v8h;
typedef __attribute__((ext_vector_type(8)))  float    v8f;
typedef __attribute__((ext_vector_type(4)))  unsigned v4u;
typedef __attribute__((ext_vector_type(8)))  int      v8i_;
typedef __attribute__((ext_vector_type(4)))  int      v4i_;

#define BB   4
#define TT   1024
#define CCC  1024
#define HHH  16
#define DDD  64
#define KTOP 256

#if defined(__HIP_DEVICE_COMPILE__) && __has_builtin(__builtin_amdgcn_tensor_load_to_lds)
#define HAVE_TDM 1
#else
#define HAVE_TDM 0
#endif

// -------- elementwise converters --------
__global__ void cvt_f32_to_f16(const float* __restrict__ in, _Float16* __restrict__ out, int n) {
  int i = blockIdx.x * blockDim.x + threadIdx.x;
  if (i < n) out[i] = (_Float16)in[i];
}

__global__ void mask_w_f16(const float* __restrict__ w, const float* __restrict__ m,
                           _Float16* __restrict__ out, int n) {
  int i = blockIdx.x * blockDim.x + threadIdx.x;
  if (i < n) out[i] = (_Float16)(w[i] * m[i]);
}

// -------- WMMA GEMM: out[bt,o] = sum_c A[bt,c] * W[o,c] + bias[o] --------
// M = 4096 (B*T), N = K = 1024.  8 waves per block, each wave owns a 16x64 tile.
__global__ __launch_bounds__(256) void gemm_wmma(
    const _Float16* __restrict__ A,     // [4096,1024] f16
    const _Float16* __restrict__ W,     // [1024,1024] f16 (masked)
    const float* __restrict__ bias,     // [1024]
    _Float16* __restrict__ outh,        // f16 output (or null)
    float* __restrict__ outf)           // f32 output (or null)
{
  const int lane = threadIdx.x & 31;
  const int wave = threadIdx.x >> 5;
  const int m0   = blockIdx.x * 16;
  const int n0   = blockIdx.y * 512 + wave * 64;
  const int col  = lane & 15;
  const int kbA  = (lane >> 4) * 16;   // A K-base per lane half
  const int kbB  = (lane >> 4) * 16;   // B K-base per lane half
  const int row  = m0 + col;           // A row for this lane (M = lane%16)

  v8f acc[4];
#pragma unroll
  for (int nt = 0; nt < 4; nt++)
#pragma unroll
    for (int r = 0; r < 8; r++) acc[nt][r] = 0.0f;

  for (int kc = 0; kc < CCC; kc += 32) {
    const _Float16* ap = A + (size_t)row * CCC + kc + kbA;
    __builtin_prefetch((const void*)(ap + 64), 0, 0);   // global_prefetch_b8, next chunks
    v8h alo = *(const v8h*)ap;
    v8h ahi = *(const v8h*)(ap + 8);
    v16h a;
#pragma unroll
    for (int j = 0; j < 8; j++) { a[j] = alo[j]; a[8 + j] = ahi[j]; }
#pragma unroll
    for (int nt = 0; nt < 4; nt++) {
      const _Float16* bp = W + (size_t)(n0 + nt * 16 + col) * CCC + kc + kbB;
      v16h b = *(const v16h*)bp;   // 32B aligned: contiguous K along c
      acc[nt] = __builtin_amdgcn_wmma_f32_16x16x32_f16(
          false, a, false, b, (short)0, acc[nt], false, false);
    }
  }

  const int rbase = (lane >> 4) * 8;
#pragma unroll
  for (int nt = 0; nt < 4; nt++) {
    int o = n0 + nt * 16 + col;
    float bv = bias[o];
#pragma unroll
    for (int r = 0; r < 8; r++) {
      int mrow = m0 + rbase + r;
      float v = acc[nt][r] + bv;
      if (outf) outf[(size_t)mrow * CCC + o] = v;
      else      outh[(size_t)mrow * CCC + o] = (_Float16)v;
    }
  }
}

// -------- TDM helpers --------
// 2-D tile load Global->LDS via Tensor Data Mover: rows x cols (f16), row stride in elems.
// Descriptor packing per CDNA5 ISA sec 8.3/8.4 (D# group0 + group1; remaining groups zero
// for a 2-D tile).  This toolchain declares the 6-arg builtin form.
__device__ __forceinline__ void tdm_load_2d(const _Float16* gsrc, void* lds_dst,
                                            unsigned rows, unsigned cols,
                                            unsigned row_stride_elems) {
#if HAVE_TDM
  unsigned lds_off = (unsigned)(size_t)lds_dst;          // LDS aperture: low 32 bits = byte offset
  unsigned long long ga = (unsigned long long)(size_t)gsrc;
  v4u g0;
  g0[0] = 1u;                                            // count=1 (valid), user mode, no gather
  g0[1] = lds_off;                                       // lds_addr
  g0[2] = (unsigned)(ga & 0xFFFFFFFFu);                  // global_addr[31:0]
  g0[3] = (unsigned)((ga >> 32) & 0x1FFFFFFu) | (2u << 30); // global_addr[56:32] | type=2
  v8i_ g1;
  g1[0] = (int)(1u << 16);                               // wg_mask=0, data_size=1 (2 bytes)
  g1[1] = (int)((cols & 0xFFFFu) << 16);                 // tensor_dim0[15:0]
  g1[2] = (int)(((cols >> 16) & 0xFFFFu) | ((rows & 0xFFFFu) << 16)); // td0 hi | td1 lo
  g1[3] = (int)(((rows >> 16) & 0xFFFFu) | ((cols & 0xFFFFu) << 16)); // td1 hi | tile_dim0
  g1[4] = (int)(rows & 0xFFFFu);                         // tile_dim1 | tile_dim2=0
  g1[5] = (int)row_stride_elems;                         // tensor_dim0_stride[31:0]
  g1[6] = 0;                                             // stride hi | td1_stride lo
  g1[7] = 0;
  v4i_ z4; z4[0] = 0; z4[1] = 0; z4[2] = 0; z4[3] = 0;
  v8i_ z8;
#pragma unroll
  for (int i = 0; i < 8; i++) z8[i] = 0;
  __builtin_amdgcn_tensor_load_to_lds(g0, g1, z4, z4, z8, 0);
#else
  (void)gsrc; (void)lds_dst; (void)rows; (void)cols; (void)row_stride_elems;
#endif
}

__device__ __forceinline__ void tdm_wait() {
#if defined(__HIP_DEVICE_COMPILE__)
#if __has_builtin(__builtin_amdgcn_s_wait_tensorcnt)
  __builtin_amdgcn_s_wait_tensorcnt((short)0);
#else
  asm volatile("s_wait_tensorcnt 0x0" ::: "memory");
#endif
#endif
}

// Fallback cooperative panel copy (1024x64 f16, src row stride CCC) when TDM is absent.
__device__ __forceinline__ void stage_copy_panel(_Float16* dst, const _Float16* src, int tid) {
#pragma unroll
  for (int i = 0; i < 32; i++) {
    int idx = tid + 256 * i;                 // v8h index, 8192 total
    const _Float16* s = src + (size_t)(idx >> 3) * CCC + (size_t)(idx & 7) * 8;
    *(v8h*)(dst + (size_t)idx * 8) = *(const v8h*)s;
  }
}

// -------- fused attention: scores -> exact top-k -> softmax(zeros) -> P*V --------
__device__ __forceinline__ unsigned sortable(float f) {
  unsigned u = __float_as_uint(f);
  return (u & 0x80000000u) ? ~u : (u | 0x80000000u);
}

__global__ __launch_bounds__(256) void brain_attn(
    const _Float16* __restrict__ qh,
    const _Float16* __restrict__ kh,
    const _Float16* __restrict__ vh,
    _Float16* __restrict__ yh)
{
  extern __shared__ char smem[];
  _Float16* kv = (_Float16*)smem;                        // [1024][64] staged K, then V (128KB)
  float*    sc = (float*)(smem + 131072);                // [16][1024] scores / exp / partials
  _Float16* pr = (_Float16*)(smem + 131072 + 65536);     // [16][1024] probs f16

  const int tile = blockIdx.x;          // query tile (t0 = tile*16)
  const int h    = blockIdx.y;
  const int b    = blockIdx.z;
  const int lane = threadIdx.x & 31;
  const int wave = threadIdx.x >> 5;
  const int t0   = tile * 16;
  const int col  = lane & 15;
  const int kbA  = (lane >> 4) * 16;
  const int kbB  = (lane >> 4) * 16;
  const int rbase = (lane >> 4) * 8;
  const size_t base = (size_t)b * TT * CCC + (size_t)h * DDD;

  // ---- stage K panel [1024 s][64 d] into LDS (TDM DMA; TENSORcnt) ----
#if HAVE_TDM
  if (wave == 0) { tdm_load_2d(kh + base, kv, TT, DDD, CCC); tdm_wait(); }
#else
  stage_copy_panel(kv, kh + base, (int)threadIdx.x);
#endif

  // Q tile for this block (registers, global: only 16x64)
  v16h aq0, aq1;
  {
    const _Float16* qp = qh + base + (size_t)(t0 + col) * CCC;
    v8h l0 = *(const v8h*)(qp + 0  + kbA);
    v8h h0 = *(const v8h*)(qp + 8  + kbA);
    v8h l1 = *(const v8h*)(qp + 32 + kbA);
    v8h h1 = *(const v8h*)(qp + 40 + kbA);
#pragma unroll
    for (int j = 0; j < 8; j++) {
      aq0[j] = l0[j]; aq0[8 + j] = h0[j];
      aq1[j] = l1[j]; aq1[8 + j] = h1[j];
    }
  }
  __syncthreads();   // K panel resident

  // ---- phase 1: S = Q Kt * (1/sqrt(D)) into LDS ----
  const float scale = 0.125f;  // 1/sqrt(64)
  for (int st = wave; st < TT / 16; st += 8) {
    int s0 = st * 16;
    const _Float16* kp = kv + (size_t)(s0 + col) * DDD;  // B[k=d][n=s] = K[s,d] from LDS
    v16h b0 = *(const v16h*)(kp + kbB);
    v16h b1 = *(const v16h*)(kp + 32 + kbB);
    v8f acc;
#pragma unroll
    for (int r = 0; r < 8; r++) acc[r] = 0.0f;
    acc = __builtin_amdgcn_wmma_f32_16x16x32_f16(false, aq0, false, b0, (short)0, acc, false, false);
    acc = __builtin_amdgcn_wmma_f32_16x16x32_f16(false, aq1, false, b1, (short)0, acc, false, false);
#pragma unroll
    for (int r = 0; r < 8; r++)
      sc[(rbase + r) * 1024 + s0 + col] = acc[r] * scale;
  }
  __syncthreads();   // scores complete; everyone done reading K panel

  // ---- kick off V panel DMA now; it overlaps the whole top-k/softmax phase ----
#if HAVE_TDM
  if (wave == 0) tdm_load_2d(vh + base, kv, TT, DDD, CCC);
#else
  stage_copy_panel(kv, vh + base, (int)threadIdx.x);
#endif

  // ---- phase 2: exact top-k threshold + scatter-into-zeros softmax ----
  for (int r = wave; r < 16; r += 8) {
    float* row = sc + r * 1024;
    float mx = -3.402823466e38f;
    for (int j = lane; j < 1024; j += 32) mx = fmaxf(mx, row[j]);
#pragma unroll
    for (int off = 16; off; off >>= 1) mx = fmaxf(mx, __shfl_xor(mx, off, 32));

    // largest u with count(sortable(x) >= u) >= KTOP  == sortable(k-th largest)
    unsigned lo = 0u, hi = 0xFFFFFFFFu;
    while (lo < hi) {
      unsigned d = hi - lo;
      unsigned mid = lo + (d >> 1) + (d & 1u);   // ceil midpoint, overflow-safe
      int cnt = 0;
      for (int j = lane; j < 1024; j += 32)
        cnt += (sortable(row[j]) >= mid) ? 1 : 0;
#pragma unroll
      for (int off = 16; off; off >>= 1) cnt += __shfl_xor(cnt, off, 32);
      if (cnt >= KTOP) lo = mid; else hi = mid - 1;
    }
    const unsigned thr = lo;

    // dropped entries become 0 before softmax (reference semantics)
    const float m = fmaxf(mx, 0.0f);
    float lsum = 0.0f;
    for (int j = lane; j < 1024; j += 32) {
      float x  = row[j];
      float sp = (sortable(x) >= thr) ? x : 0.0f;
      float e  = __expf(sp - m);
      row[j] = e;
      lsum += e;
    }
#pragma unroll
    for (int off = 16; off; off >>= 1) lsum += __shfl_xor(lsum, off, 32);
    float inv = 1.0f / lsum;
    _Float16* prow = pr + r * 1024;
    for (int j = lane; j < 1024; j += 32) prow[j] = (_Float16)(row[j] * inv);
  }

#if HAVE_TDM
  if (wave == 0) tdm_wait();     // V panel resident
#endif
  __syncthreads();

  // ---- phase 3: Y = P (16x1024) * V (1024x64); all 8 waves (split-K by 2) ----
  const int d0 = (wave & 3) * 16;
  const int k0 = (wave >> 2) * 512;
  v8f acc;
#pragma unroll
  for (int r = 0; r < 8; r++) acc[r] = 0.0f;
  for (int kc = k0; kc < k0 + 512; kc += 32) {
    const _Float16* ap = pr + col * 1024 + kc + kbA;
    v8h alo = *(const v8h*)ap;
    v8h ahi = *(const v8h*)(ap + 8);
    v16h a;
#pragma unroll
    for (int j = 0; j < 8; j++) { a[j] = alo[j]; a[8 + j] = ahi[j]; }
    // B[k=s][n=d] = V[s, d0+n] from the LDS V panel
    const _Float16* vp = kv + (size_t)(kc + kbB) * DDD + d0 + col;
    v16h bv;
#pragma unroll
    for (int e = 0; e < 16; e++) bv[e] = vp[(size_t)e * DDD];
    acc = __builtin_amdgcn_wmma_f32_16x16x32_f16(false, a, false, bv, (short)0, acc, false, false);
  }
  // split-K reduction through LDS (reuse score region, now dead)
  float* sred = sc;
  if (wave >= 4) {
#pragma unroll
    for (int r = 0; r < 8; r++)
      sred[(wave - 4) * 256 + (rbase + r) * 16 + col] = acc[r];
  }
  __syncthreads();
  if (wave < 4) {
#pragma unroll
    for (int r = 0; r < 8; r++) {
      float v = acc[r] + sred[wave * 256 + (rbase + r) * 16 + col];
      yh[(size_t)((size_t)b * TT + t0 + rbase + r) * CCC + h * DDD + d0 + col] = (_Float16)v;
    }
  }
}

extern "C" void kernel_launch(void* const* d_in, const int* in_sizes, int n_in,
                              void* d_out, int out_size, void* d_ws, size_t ws_size,
                              hipStream_t stream) {
  (void)in_sizes; (void)n_in; (void)out_size; (void)ws_size;
  const float* x  = (const float*)d_in[0];
  const float* wq = (const float*)d_in[1];
  const float* bq = (const float*)d_in[2];
  const float* mq = (const float*)d_in[3];
  const float* wk = (const float*)d_in[4];
  const float* bk = (const float*)d_in[5];
  const float* mk = (const float*)d_in[6];
  const float* wv = (const float*)d_in[7];
  const float* bv = (const float*)d_in[8];
  const float* mv = (const float*)d_in[9];
  const float* wo = (const float*)d_in[10];
  const float* bo = (const float*)d_in[11];
  const float* mo = (const float*)d_in[12];

  // workspace layout (f16 elements): 48 MB total
  _Float16* p   = (_Float16*)d_ws;
  _Float16* xh  = p; p += (size_t)BB * TT * CCC;
  _Float16* wqh = p; p += (size_t)CCC * CCC;
  _Float16* wkh = p; p += (size_t)CCC * CCC;
  _Float16* wvh = p; p += (size_t)CCC * CCC;
  _Float16* woh = p; p += (size_t)CCC * CCC;
  _Float16* qh  = p; p += (size_t)BB * TT * CCC;
  _Float16* kh  = p; p += (size_t)BB * TT * CCC;
  _Float16* vh  = p; p += (size_t)BB * TT * CCC;
  _Float16* yh  = p; p += (size_t)BB * TT * CCC;

  const int nx = BB * TT * CCC;
  const int nw = CCC * CCC;
  cvt_f32_to_f16<<<(nx + 255) / 256, 256, 0, stream>>>(x, xh, nx);
  mask_w_f16<<<(nw + 255) / 256, 256, 0, stream>>>(wq, mq, wqh, nw);
  mask_w_f16<<<(nw + 255) / 256, 256, 0, stream>>>(wk, mk, wkh, nw);
  mask_w_f16<<<(nw + 255) / 256, 256, 0, stream>>>(wv, mv, wvh, nw);
  mask_w_f16<<<(nw + 255) / 256, 256, 0, stream>>>(wo, mo, woh, nw);

  dim3 gg((BB * TT) / 16, CCC / 512);   // (256, 2)
  gemm_wmma<<<gg, 256, 0, stream>>>(xh, wqh, bq, qh, nullptr);
  gemm_wmma<<<gg, 256, 0, stream>>>(xh, wkh, bk, kh, nullptr);
  gemm_wmma<<<gg, 256, 0, stream>>>(xh, wvh, bv, vh, nullptr);

  dim3 ga(TT / 16, HHH, BB);            // (64, 16, 4)
  size_t lds = 131072 + 65536 + 32768;  // 224 KB: V/K panel + scores + probs
  brain_attn<<<ga, 256, lds, stream>>>(qh, kh, vh, yh);

  gemm_wmma<<<gg, 256, 0, stream>>>(yh, woh, bo, nullptr, (float*)d_out);
}
